// GNNFeatureExtractor_25821343383965
// MI455X (gfx1250) — compile-verified, matrix operand order checked by reading, not run on hardware
//
#include <hip/hip_runtime.h>

// GCN feature extractor for MI455X (gfx1250, wave32, WMMA).
//   h   = x @ W1                                   (bf16 WMMA, f32 acc)
//   agg = sym-normalized scatter-add over edges    (f32 atomics in L2)
//   out = relu(agg + h/deg + b1) @ W2 + b2         (bf16 WMMA, fused epilogue)
//
// Roofline: ~200 MB of true HBM traffic @ 23.3 TB/s (~9 us floor); the edge
// phase (h/agg = 2x25.6 MB, both resident in the 192 MB L2) is atomic-bound.
// GEMM FLOPs (~4 GFLOP) are negligible, so each wave computes a full 16x64
// tile (A loaded once, 4 pre-packed B fragments, 4 accumulators) to keep the
// x read at exactly 1x HBM.

typedef __attribute__((ext_vector_type(16))) __bf16 v16bf;
typedef __attribute__((ext_vector_type(8)))  float  v8f;

#define F_IN 256
#define D    64

#define WMMA_BF16(A, B, C) \
  __builtin_amdgcn_wmma_f32_16x16x32_bf16(false, (A), false, (B), (short)0, (C), false, false)

// No-return fp32 global atomic add (GLOBAL_ATOMIC_ADD_F32, device scope ->
// RMW resolves at L2; fire-and-forget, STOREcnt tracked).
__device__ __forceinline__ void atomic_add_f32(float* p, float v) {
  asm volatile("global_atomic_add_f32 %0, %1, off scope:SCOPE_DEV"
               :: "v"(p), "v"(v) : "memory");
}

// Pack two float4 pairs into one 16-wide bf16 A/B fragment.
__device__ __forceinline__ v16bf pack_frag(float4 f0, float4 f1, float4 f2, float4 f3) {
  v16bf a;
  a[0]  = (__bf16)f0.x; a[1]  = (__bf16)f0.y; a[2]  = (__bf16)f0.z; a[3]  = (__bf16)f0.w;
  a[4]  = (__bf16)f1.x; a[5]  = (__bf16)f1.y; a[6]  = (__bf16)f1.z; a[7]  = (__bf16)f1.w;
  a[8]  = (__bf16)f2.x; a[9]  = (__bf16)f2.y; a[10] = (__bf16)f2.z; a[11] = (__bf16)f2.w;
  a[12] = (__bf16)f3.x; a[13] = (__bf16)f3.y; a[14] = (__bf16)f3.z; a[15] = (__bf16)f3.w;
  return a;
}

// ---------------- init: deg = 1 (self loop), agg = 0 --------------------
__global__ void init_kernel(float* __restrict__ deg, float* __restrict__ agg, int N) {
  long long i = (long long)blockIdx.x * blockDim.x + threadIdx.x;
  long long tot = (long long)N * D;
  if (i < tot) agg[i] = 0.0f;
  if (i < N)   deg[i] = 1.0f;
}

// ------- pack W1/W2 (f32 row-major KxD) into bf16 WMMA B fragments ------
// Fragment f = (kchunk*4 + colgroup); lane holds 16 contiguous bf16 at
// (f*32 + lane)*16: b[i] = W[(k0+half*8+i)*D + n], b[i+8] = +16 in K.
__global__ void pack_weights_kernel(const float* __restrict__ W1,
                                    const float* __restrict__ W2,
                                    __bf16* __restrict__ w1f,
                                    __bf16* __restrict__ w2f) {
  const int t = blockIdx.x * blockDim.x + threadIdx.x;
  const int n1 = (F_IN / 32) * 4 * 32;   // 1024 lane-slots for W1
  const int n2 = (D / 32) * 4 * 32;      //  256 lane-slots for W2
  const float* W; __bf16* dstbuf; int idx;
  if (t < n1)           { W = W1; dstbuf = w1f; idx = t; }
  else if (t < n1 + n2) { W = W2; dstbuf = w2f; idx = t - n1; }
  else return;
  const int lane  = idx & 31;
  const int wv    = (idx >> 5) & 3;
  const int chunk = idx >> 7;
  const int half  = lane >> 4;
  const int n     = wv * 16 + (lane & 15);
  const int k0    = chunk * 32 + half * 8;
  __bf16* dst = dstbuf + (size_t)idx * 16;
#pragma unroll
  for (int i = 0; i < 8; ++i) {
    dst[i]     = (__bf16)W[(k0 + i) * D + n];
    dst[i + 8] = (__bf16)W[(k0 + 16 + i) * D + n];
  }
}

// ---------------- degree: deg[dst] += 1 per edge ------------------------
__global__ void degree_kernel(const long long* __restrict__ ei,
                              float* __restrict__ deg, int E) {
  int e = blockIdx.x * blockDim.x + threadIdx.x;
  if (e < E) {
    long long dst = ei[(long long)E + e];
    atomic_add_f32(&deg[dst], 1.0f);
  }
}

// ---------------- gemm1: h = x @ W1 -------------------------------------
// One wave32 per 16x64 output tile: A fragment loaded once per K-chunk
// (4x global_load_b128), 4 pre-packed B fragments, 4 WMMA accumulators.
__global__ void gemm1_kernel(const float* __restrict__ x,
                             const __bf16* __restrict__ w1f,
                             float* __restrict__ h, int N) {
  const int lane = threadIdx.x & 31;
  const int wid  = threadIdx.x >> 5;
  const int half = lane >> 4;
  const int mrow = lane & 15;
  const int tile = blockIdx.x * (blockDim.x >> 5) + wid;
  const int row0 = tile * 16;
  if (row0 >= N) return;
  const int row  = row0 + mrow;
  const float* xr = x + (long long)(row < N ? row : N - 1) * F_IN;

  v8f c0 = {}, c1 = {}, c2 = {}, c3 = {};
  for (int k0 = 0; k0 < F_IN; k0 += 32) {
    const int ka = k0 + half * 8;
    if (k0 + 32 < F_IN) __builtin_prefetch(xr + ka + 32, 0, 0);  // global_prefetch_b8
    const float4 f0 = *(const float4*)(xr + ka);
    const float4 f1 = *(const float4*)(xr + ka + 4);
    const float4 f2 = *(const float4*)(xr + ka + 16);
    const float4 f3 = *(const float4*)(xr + ka + 20);
    const v16bf a = pack_frag(f0, f1, f2, f3);
    const __bf16* fb = w1f + ((size_t)(k0 >> 5) * 4 * 32 + lane) * 16;
    const v16bf b0 = *(const v16bf*)(fb);
    const v16bf b1 = *(const v16bf*)(fb + 32 * 16);
    const v16bf b2 = *(const v16bf*)(fb + 64 * 16);
    const v16bf b3 = *(const v16bf*)(fb + 96 * 16);
    c0 = WMMA_BF16(a, b0, c0);
    c1 = WMMA_BF16(a, b1, c1);
    c2 = WMMA_BF16(a, b2, c2);
    c3 = WMMA_BF16(a, b3, c3);
  }
  // D layout: element r -> row r + 8*half, column = colgroup*16 + (lane&15)
  if (row0 + 16 <= N) {              // uniform fast path: no exec juggling
#pragma unroll
    for (int r = 0; r < 8; ++r) {
      float* orow = h + (long long)(row0 + r + half * 8) * D + mrow;
      orow[0]  = c0[r]; orow[16] = c1[r]; orow[32] = c2[r]; orow[48] = c3[r];
    }
  } else {
#pragma unroll
    for (int r = 0; r < 8; ++r) {
      const int rr = row0 + r + half * 8;
      if (rr < N) {
        float* orow = h + (long long)rr * D + mrow;
        orow[0]  = c0[r]; orow[16] = c1[r]; orow[32] = c2[r]; orow[48] = c3[r];
      }
    }
  }
}

// ---------------- edge aggregation: one wave32 per edge -----------------
__global__ void edge_kernel(const long long* __restrict__ ei,
                            const float* __restrict__ deg,
                            const float* __restrict__ h,
                            float* __restrict__ agg, int E) {
  const int gw   = (int)(((long long)blockIdx.x * blockDim.x + threadIdx.x) >> 5);
  const int lane = threadIdx.x & 31;
  if (gw >= E) return;
  const long long j = ei[gw];                  // source
  const long long i = ei[(long long)E + gw];   // target
  const float norm = rsqrtf(deg[i]) * rsqrtf(deg[j]);
  const float2 v = *(const float2*)(h + j * D + lane * 2);
  float* p = agg + i * D + lane * 2;
  atomic_add_f32(p,     norm * v.x);
  atomic_add_f32(p + 1, norm * v.y);
}

// ---- gemm2: out = relu(agg + h/deg + b1) @ W2 + b2 (fused epilogue) ----
__global__ void gemm2_kernel(const float* __restrict__ agg,
                             const float* __restrict__ hbuf,
                             const float* __restrict__ deg,
                             const float* __restrict__ b1,
                             const __bf16* __restrict__ w2f,
                             const float* __restrict__ b2,
                             float* __restrict__ out, int N) {
  const int lane = threadIdx.x & 31;
  const int wid  = threadIdx.x >> 5;
  const int half = lane >> 4;
  const int mrow = lane & 15;
  const int tile = blockIdx.x * (blockDim.x >> 5) + wid;
  const int row0 = tile * 16;
  if (row0 >= N) return;
  const int row  = row0 + mrow;
  const long long rb = (long long)(row < N ? row : N - 1) * D;
  const float invd = 1.0f / deg[row < N ? row : N - 1];

  v8f c0 = {}, c1 = {}, c2 = {}, c3 = {};
#pragma unroll
  for (int k0 = 0; k0 < D; k0 += 32) {
    const int ka = k0 + half * 8;
    const float4 g0 = *(const float4*)(agg + rb + ka);
    const float4 g1 = *(const float4*)(agg + rb + ka + 4);
    const float4 g2 = *(const float4*)(agg + rb + ka + 16);
    const float4 g3 = *(const float4*)(agg + rb + ka + 20);
    const float4 s0 = *(const float4*)(hbuf + rb + ka);
    const float4 s1 = *(const float4*)(hbuf + rb + ka + 4);
    const float4 s2 = *(const float4*)(hbuf + rb + ka + 16);
    const float4 s3 = *(const float4*)(hbuf + rb + ka + 20);
    const float4 t0 = *(const float4*)(b1 + ka);
    const float4 t1 = *(const float4*)(b1 + ka + 4);
    const float4 t2 = *(const float4*)(b1 + ka + 16);
    const float4 t3 = *(const float4*)(b1 + ka + 20);
    float4 a0, a1, a2, a3;
    a0.x = fmaxf(g0.x + s0.x * invd + t0.x, 0.f); a0.y = fmaxf(g0.y + s0.y * invd + t0.y, 0.f);
    a0.z = fmaxf(g0.z + s0.z * invd + t0.z, 0.f); a0.w = fmaxf(g0.w + s0.w * invd + t0.w, 0.f);
    a1.x = fmaxf(g1.x + s1.x * invd + t1.x, 0.f); a1.y = fmaxf(g1.y + s1.y * invd + t1.y, 0.f);
    a1.z = fmaxf(g1.z + s1.z * invd + t1.z, 0.f); a1.w = fmaxf(g1.w + s1.w * invd + t1.w, 0.f);
    a2.x = fmaxf(g2.x + s2.x * invd + t2.x, 0.f); a2.y = fmaxf(g2.y + s2.y * invd + t2.y, 0.f);
    a2.z = fmaxf(g2.z + s2.z * invd + t2.z, 0.f); a2.w = fmaxf(g2.w + s2.w * invd + t2.w, 0.f);
    a3.x = fmaxf(g3.x + s3.x * invd + t3.x, 0.f); a3.y = fmaxf(g3.y + s3.y * invd + t3.y, 0.f);
    a3.z = fmaxf(g3.z + s3.z * invd + t3.z, 0.f); a3.w = fmaxf(g3.w + s3.w * invd + t3.w, 0.f);
    const v16bf a = pack_frag(a0, a1, a2, a3);
    const __bf16* fb = w2f + ((size_t)(k0 >> 5) * 4 * 32 + lane) * 16;
    const v16bf b0 = *(const v16bf*)(fb);
    const v16bf b1v = *(const v16bf*)(fb + 32 * 16);
    const v16bf b2v = *(const v16bf*)(fb + 64 * 16);
    const v16bf b3v = *(const v16bf*)(fb + 96 * 16);
    c0 = WMMA_BF16(a, b0,  c0);
    c1 = WMMA_BF16(a, b1v, c1);
    c2 = WMMA_BF16(a, b2v, c2);
    c3 = WMMA_BF16(a, b3v, c3);
  }
  const float z0 = b2[mrow], z1 = b2[16 + mrow], z2 = b2[32 + mrow], z3 = b2[48 + mrow];
  if (row0 + 16 <= N) {
#pragma unroll
    for (int r = 0; r < 8; ++r) {
      float* orow = out + (long long)(row0 + r + half * 8) * D + mrow;
      orow[0]  = c0[r] + z0; orow[16] = c1[r] + z1;
      orow[32] = c2[r] + z2; orow[48] = c3[r] + z3;
    }
  } else {
#pragma unroll
    for (int r = 0; r < 8; ++r) {
      const int rr = row0 + r + half * 8;
      if (rr < N) {
        float* orow = out + (long long)rr * D + mrow;
        orow[0]  = c0[r] + z0; orow[16] = c1[r] + z1;
        orow[32] = c2[r] + z2; orow[48] = c3[r] + z3;
      }
    }
  }
}

extern "C" void kernel_launch(void* const* d_in, const int* in_sizes, int n_in,
                              void* d_out, int out_size, void* d_ws, size_t ws_size,
                              hipStream_t stream) {
  const float*     x  = (const float*)d_in[0];
  const long long* ei = (const long long*)d_in[1];   // int64 (2, E)
  const float*     W1 = (const float*)d_in[2];
  const float*     b1 = (const float*)d_in[3];
  const float*     W2 = (const float*)d_in[4];
  const float*     b2 = (const float*)d_in[5];
  float* out = (float*)d_out;

  const int N = in_sizes[0] / F_IN;
  const int E = in_sizes[1] / 2;

  // Workspace: deg[N pad] | h[N*D] | agg[N*D] | w1f (32KB bf16) | w2f (8KB)
  float* deg = (float*)d_ws;
  const long long degPad = ((long long)N + 4095) / 4096 * 4096;
  float*  h   = deg + degPad;
  float*  agg = h + (long long)N * D;
  __bf16* w1f = (__bf16*)(agg + (long long)N * D);
  __bf16* w2f = w1f + (size_t)F_IN * D;

  const long long tot = (long long)N * D;
  init_kernel<<<(int)((tot + 255) / 256), 256, 0, stream>>>(deg, agg, N);
  pack_weights_kernel<<<5, 256, 0, stream>>>(W1, W2, w1f, w2f);
  degree_kernel<<<(E + 255) / 256, 256, 0, stream>>>(ei, deg, E);

  const int ntiles = (N + 15) / 16;              // one wave per tile
  const int wavesPerBlock = 8;                   // 256 threads
  const int gblocks = (ntiles + wavesPerBlock - 1) / wavesPerBlock;
  gemm1_kernel<<<gblocks, 32 * wavesPerBlock, 0, stream>>>(x, w1f, h, N);

  edge_kernel<<<(E + 7) / 8, 256, 0, stream>>>(ei, deg, h, agg, E);

  gemm2_kernel<<<gblocks, 32 * wavesPerBlock, 0, stream>>>(agg, h, deg, b1, w2f, b2, out, N);
}
